// mpNet_Constrained_76287209112058
// MI455X (gfx1250) — matrix operand order repaired and checked by reading, not compiled, not run on hardware
//
#include <hip/hip_runtime.h>
#include <hip/hip_bf16.h>
#include <math.h>

// ---------------------------------------------------------------------------
// mpNet constrained matching pursuit, fused for MI455X (gfx1250, wave32, WMMA)
//   Hot loop: c = residual @ conj(D_M) -> argmax|c| per row -> rank-1 update,
//   all k iterations fused in one kernel; f16 dictionary L2-resident; complex
//   GEMM as two real WMMA streams (composite K=256), explicit software
//   pipelining of B fragments (double-buffered in registers, branch-free).
//   CDNA5 paths: v_wmma_f32_16x16x32_f16,
//                global_load_async_to_lds_b128 + s_wait_asynccnt (k_dm),
//                tensor_load_to_lds + s_wait_tensorcnt (k_mp residual stage).
// ---------------------------------------------------------------------------

typedef __attribute__((ext_vector_type(16))) _Float16 v16h;
typedef __attribute__((ext_vector_type(8)))  float    v8f;
typedef __attribute__((ext_vector_type(4)))  unsigned int v4u;
typedef __attribute__((ext_vector_type(4)))  int      v4i;
typedef __attribute__((ext_vector_type(8)))  int      v8i;

union Frag16 { v16h v; _Float16 h[16]; unsigned long long u[4]; };

#define N_ANT   256
#define M_MEAS  128
#define A_GRID  4096
#define N_SAMP  2048
#define LAM     0.01
#define PI_D    3.14159265358979323846

#ifndef __has_builtin
#define __has_builtin(x) 0
#endif

#if __has_builtin(__builtin_amdgcn_global_load_async_to_lds_b128) && \
    __has_builtin(__builtin_amdgcn_s_wait_asynccnt)
#define USE_ASYNC_LDS 1
#else
#define USE_ASYNC_LDS 0
#endif

#if __has_builtin(__builtin_amdgcn_tensor_load_to_lds) && \
    __has_builtin(__builtin_amdgcn_s_wait_tensorcnt)
#define USE_TDM 1
#else
#define USE_TDM 0
#endif

// async-to-LDS builtin takes typed b128 (int4) pointers: global AS1 src, LDS AS3 dst
#define ASYNC_G(p) ((__attribute__((address_space(1))) v4i*)(p))
#define ASYNC_L(p) ((__attribute__((address_space(3))) v4i*)(p))

// ---------------- Kernel 1: steering matrix W[a][n] (atom-major, interleaved)
__global__ __launch_bounds__(256) void k_W(const double* __restrict__ pos,
                                           const double* __restrict__ doa,
                                           const double* __restrict__ g,
                                           float* __restrict__ W) {
  int gid = blockIdx.x * 256 + threadIdx.x;      // 4096*256 threads
  int a = gid >> 8;
  int n = gid & 255;
  double phase = (-2.0 * PI_D / LAM) * pos[n] * sin(doa[a]);
  double s, c;
  sincos(phase, &s, &c);
  double gv = g[a];
  W[(size_t)a * 512 + n * 2]     = (float)(gv * c);
  W[(size_t)a * 512 + n * 2 + 1] = (float)(gv * s);
}

// ---------------- Kernel 2: D_M[m][a] = sum_n conj(M[n][m]) * W[n][a]
// W column staged into LDS; async double-buffered when the toolchain has the
// CDNA5 async-to-LDS builtins (stage column j+1 while computing column j).
__global__ __launch_bounds__(128) void k_dm(const double* __restrict__ M,
                                            const float* __restrict__ W,
                                            float* __restrict__ DM,
                                            float* __restrict__ norm2) {
  __shared__ __align__(16) float Wl[2][512];     // two W columns (256 complex)
  int m = threadIdx.x;
  float nAcc = 0.f;

#if USE_ASYNC_LDS
  {
    int a0 = blockIdx.x * 8;
    __builtin_amdgcn_global_load_async_to_lds_b128(
        ASYNC_G(W + (size_t)a0 * 512 + m * 4), ASYNC_L(&Wl[0][m * 4]), 0, 0);
  }
#endif

  for (int j = 0; j < 8; ++j) {
    int a = blockIdx.x * 8 + j;
    const float* Wc;
#if USE_ASYNC_LDS
    if (j + 1 < 8) {
      __builtin_amdgcn_global_load_async_to_lds_b128(
          ASYNC_G(W + (size_t)(a + 1) * 512 + m * 4),
          ASYNC_L(&Wl[(j + 1) & 1][m * 4]), 0, 0);
      __builtin_amdgcn_s_wait_asynccnt(1);       // buffer j complete
    } else {
      __builtin_amdgcn_s_wait_asynccnt(0);
    }
    __syncthreads();
    Wc = Wl[j & 1];
#else
    __syncthreads();
    ((float4*)Wl[0])[m] = ((const float4*)(W + (size_t)a * 512))[m];
    __syncthreads();
    Wc = Wl[0];
#endif
    float ar = 0.f, ai = 0.f;
    for (int n = 0; n < 256; ++n) {
      float mr = (float)M[((size_t)n * 128 + m) * 2];
      float mi = (float)M[((size_t)n * 128 + m) * 2 + 1];
      float wr = Wc[n * 2], wi = Wc[n * 2 + 1];
      ar += mr * wr + mi * wi;                   // conj(M) * W
      ai += mr * wi - mi * wr;
    }
    DM[((size_t)m * 4096 + a) * 2]     = ar;
    DM[((size_t)m * 4096 + a) * 2 + 1] = ai;
    nAcc += ar * ar + ai * ai;
    __syncthreads();                             // done reading buf before restage
  }
  atomicAdd(&norm2[m], nAcc);                    // row-norm^2 over atoms
}

// ---------------- Kernel 3: normalize rows; pack WMMA B operands (f16) + Dt
// Bre[a][k]: k<128 -> Re(d[k]),  k>=128 -> Im(d[k-128])      (c_re = A.Bre)
// Bim[a][k]: k<128 -> -Im(d[k]), k>=128 -> Re(d[k-128])      (c_im = A.Bim)
__global__ __launch_bounds__(256) void k_pack(const float* __restrict__ DM,
                                              const float* __restrict__ norm2,
                                              float* __restrict__ Dt,
                                              _Float16* __restrict__ Bre,
                                              _Float16* __restrict__ Bim) {
  int m = blockIdx.x;                            // 0..127
  float inv = rsqrtf(norm2[m] + 1e-30f);
  for (int a = threadIdx.x; a < 4096; a += 256) {
    float dre = DM[((size_t)m * 4096 + a) * 2] * inv;
    float dim = DM[((size_t)m * 4096 + a) * 2 + 1] * inv;
    Dt[((size_t)a * 128 + m) * 2]     = dre;
    Dt[((size_t)a * 128 + m) * 2 + 1] = dim;
    Bre[(size_t)a * 256 + m]       = (_Float16)dre;
    Bre[(size_t)a * 256 + 128 + m] = (_Float16)dim;
    Bim[(size_t)a * 256 + m]       = (_Float16)(-dim);
    Bim[(size_t)a * 256 + 128 + m] = (_Float16)dre;
  }
}

// load one B fragment pair (Bre/Bim) at half-element offset `base`
#define LOADB(dstR, dstI, base) do {                                        \
    const unsigned long long* pr_ = (const unsigned long long*)(Bre + (base)); \
    const unsigned long long* pi_ = (const unsigned long long*)(Bim + (base)); \
    (dstR).u[0] = pr_[0]; (dstR).u[1] = pr_[1];                             \
    (dstR).u[2] = pr_[2]; (dstR).u[3] = pr_[3];                             \
    (dstI).u[0] = pi_[0]; (dstI).u[1] = pi_[1];                             \
    (dstI).u[2] = pi_[2]; (dstI).u[3] = pi_[3];                             \
  } while (0)

// ---------------- Kernel 4: fused matching pursuit (all k iterations)
// 1 workgroup = 16 sample rows, 8 wave32s; residual resident in LDS.
__global__ __launch_bounds__(256) void k_mp(const double* __restrict__ xM,
                                            const int* __restrict__ kptr,
                                            const float* __restrict__ Dt,
                                            const _Float16* __restrict__ Bre,
                                            const _Float16* __restrict__ Bim,
                                            float* __restrict__ out) {
  __shared__ float resR[16][128];
  __shared__ float resI[16][128];
  __shared__ __align__(16) _Float16 Acomp[16][256];   // [Re | Im] composite, K=256
  __shared__ float wbMag[8][16];
  __shared__ int   wbIdx[8][16];
  __shared__ float wbCre[8][16];
  __shared__ float wbCim[8][16];
  __shared__ int   rbIdx[16];
  __shared__ float rbCre[16];
  __shared__ float rbCim[16];
#if USE_TDM
  __shared__ __align__(16) double stage[16][256];     // 16 rows x 128 complex128
#endif

  const int tid  = threadIdx.x;
  const int lane = tid & 31;
  const int w    = tid >> 5;
  const int row0 = blockIdx.x * 16;

#if USE_TDM
  // TDM: DMA the 16x256-double x_M tile into LDS (2D tensor descriptor).
  if (w == 0) {
    unsigned lds_off = (unsigned)(unsigned long long)
        (__attribute__((address_space(3))) char*)&stage[0][0];
    unsigned long long ga =
        (unsigned long long)(const char*)(xM + (size_t)row0 * 256);
    v4u g0;
    g0[0] = 1u;                                   // count=1 (valid), user mode
    g0[1] = lds_off;                              // lds_addr
    g0[2] = (unsigned)(ga & 0xFFFFFFFFu);         // global_addr[31:0]
    g0[3] = (unsigned)((ga >> 32) & 0x01FFFFFFu)  // global_addr[56:32]
            | (2u << 30);                         // type=2 ("image")
    v8i g1;
    g1[0] = (int)(3u << 16);                      // data_size=8B, no multicast
    g1[1] = (int)(256u << 16);                    // tensor_dim0[15:0]=256 (8B units)
    g1[2] = (int)(2048u << 16);                   // tensor_dim1[15:0]=2048 rows
    g1[3] = (int)(256u << 16);                    // tile_dim0=256
    g1[4] = 16;                                   // tile_dim1=16 rows
    g1[5] = 256;                                  // tensor_dim0_stride=256
    g1[6] = 0;
    g1[7] = 0;
    v4i gz = {0, 0, 0, 0};
#if defined(__clang_major__) && __clang_major__ >= 23
    v8i gz8 = {0, 0, 0, 0, 0, 0, 0, 0};
    __builtin_amdgcn_tensor_load_to_lds(g0, g1, gz, gz, gz8, 0);
#else
    __builtin_amdgcn_tensor_load_to_lds(g0, g1, gz, gz, 0);
#endif
    __builtin_amdgcn_s_wait_tensorcnt(0);
  }
  __syncthreads();
  for (int e = tid; e < 2048; e += 256) {
    int r = e >> 7, m = e & 127;
    resR[r][m] = (float)stage[r][2 * m];
    resI[r][m] = (float)stage[r][2 * m + 1];
  }
#else
  for (int e = tid; e < 2048; e += 256) {
    int r = e >> 7, m = e & 127;
    size_t gpos = ((size_t)(row0 + r) * 128 + m) * 2;
    resR[r][m] = (float)xM[gpos];
    resI[r][m] = (float)xM[gpos + 1];
  }
#endif

  const int K      = kptr[0];
  const int arow   = lane & 15;        // A: M row (per ISA A layout)
  const int half   = lane >> 4;        // 0: lanes 0-15, 1: lanes 16-31
  const int batom  = lane & 15;        // B/C/D: N column = atom within tile
  const int rbase  = half * 8;         // C/D: rows rbase..rbase+7 in this lane
  const int a0base = w * 512;          // each wave scans 512 atoms (32 tiles)

  for (int it = 0; it < K; ++it) {
    __syncthreads();
    // rebuild f16 composite A from residual
    for (int e = tid; e < 2048; e += 256) {
      int r = e >> 7, m = e & 127;
      Acomp[r][m]       = (_Float16)resR[r][m];
      Acomp[r][128 + m] = (_Float16)resI[r][m];
    }
    __syncthreads();

    // A fragments for 8 K-chunks of 32 (contiguous layout: lanes<16 hold K
    // {kc..kc+15}; lanes>=16 hold {kc+16..kc+31})
    Frag16 af[8];
#pragma unroll
    for (int c = 0; c < 8; ++c) {
      int o1 = c * 32 + half * 16;
      const unsigned long long* s0 = (const unsigned long long*)&Acomp[arow][o1];
      const unsigned long long* s1 = (const unsigned long long*)&Acomp[arow][o1 + 8];
      af[c].u[0] = s0[0]; af[c].u[1] = s0[1];
      af[c].u[2] = s1[0]; af[c].u[3] = s1[1];
    }

    float bMag[8], bCre[8], bCim[8]; int bIdx[8];
#pragma unroll
    for (int r = 0; r < 8; ++r) { bMag[r] = -1.0f; bIdx[r] = 0; bCre[r] = 0.f; bCim[r] = 0.f; }

    // explicit software pipeline: B fragments for step s+1 are issued before
    // the WMMAs of step s; always-load (last step re-reads bb, data unused)
    // keeps the loop branch-free and every fragment defined.
    size_t bb = (size_t)(a0base + batom) * 256 + half * 16;
    Frag16 fr, fi, frN, fiN;
    LOADB(fr, fi, bb);
    for (int t = 0; t < 32; ++t) {
      v8f cre = {};
      v8f cim = {};
#pragma unroll
      for (int c = 0; c < 8; ++c) {
        size_t nb;
        if (c + 1 < 8) {
          nb = bb + (size_t)(c + 1) * 32;             // next chunk, same tile
        } else {
          nb = (t + 1 < 32) ? bb + (size_t)16 * 256   // next tile, chunk 0
                            : bb;                     // harmless dummy (unused)
        }
        LOADB(frN, fiN, nb);
        cre = __builtin_amdgcn_wmma_f32_16x16x32_f16(false, af[c].v, false, fr.v,
                                                     (short)0, cre, false, false);
        cim = __builtin_amdgcn_wmma_f32_16x16x32_f16(false, af[c].v, false, fi.v,
                                                     (short)0, cim, false, false);
        fr = frN; fi = fiN;
      }
      union { v8f v; float f[8]; } ur, ui;
      ur.v = cre; ui.v = cim;
      int atom = a0base + t * 16 + batom;
#pragma unroll
      for (int r = 0; r < 8; ++r) {
        float m2 = ur.f[r] * ur.f[r] + ui.f[r] * ui.f[r];
        if (m2 > bMag[r]) { bMag[r] = m2; bIdx[r] = atom; bCre[r] = ur.f[r]; bCim[r] = ui.f[r]; }
      }
      bb += (size_t)16 * 256;
    }

    // reduce across the 16 lanes of each half (rows rbase..rbase+7)
#pragma unroll
    for (int d = 8; d >= 1; d >>= 1) {
#pragma unroll
      for (int r = 0; r < 8; ++r) {
        float oM = __shfl_xor(bMag[r], d);
        int   oI = __shfl_xor(bIdx[r], d);
        float oR = __shfl_xor(bCre[r], d);
        float oC = __shfl_xor(bCim[r], d);
        bool take = (oM > bMag[r]) || ((oM == bMag[r]) && (oI < bIdx[r]));
        if (take) { bMag[r] = oM; bIdx[r] = oI; bCre[r] = oR; bCim[r] = oC; }
      }
    }
    if ((lane & 15) == 0) {
#pragma unroll
      for (int r = 0; r < 8; ++r) {
        wbMag[w][rbase + r] = bMag[r];
        wbIdx[w][rbase + r] = bIdx[r];
        wbCre[w][rbase + r] = bCre[r];
        wbCim[w][rbase + r] = bCim[r];
      }
    }
    __syncthreads();
    if (tid < 16) {                        // cross-wave argmax per row
      float bm = -2.f; int bi = 0; float br = 0.f, bc = 0.f;
      for (int ww = 0; ww < 8; ++ww) {
        float m2 = wbMag[ww][tid]; int ii = wbIdx[ww][tid];
        if (m2 > bm || (m2 == bm && ii < bi)) { bm = m2; bi = ii; br = wbCre[ww][tid]; bc = wbCim[ww][tid]; }
      }
      rbIdx[tid] = bi; rbCre[tid] = br; rbCim[tid] = bc;
    }
    __syncthreads();
    // residual -= c * D_M[:, idx]
    for (int e = tid; e < 2048; e += 256) {
      int r = e >> 7, m = e & 127;
      int idx = rbIdx[r]; float cr = rbCre[r], ci = rbCim[r];
      float dre = Dt[((size_t)idx * 128 + m) * 2];
      float dim = Dt[((size_t)idx * 128 + m) * 2 + 1];
      resR[r][m] -= cr * dre - ci * dim;
      resI[r][m] -= cr * dim + ci * dre;
    }
  }
  __syncthreads();

  const size_t H = (size_t)N_SAMP * M_MEAS * 2;   // residual block size (floats)
  for (int e = tid; e < 2048; e += 256) {
    int r = e >> 7, m = e & 127;
    size_t gpos = ((size_t)(row0 + r) * 128 + m) * 2;
    float rr = resR[r][m], ri = resI[r][m];
    out[gpos]     = rr;                           // residual_M
    out[gpos + 1] = ri;
    out[H + gpos]     = (float)xM[gpos]     - rr; // h_hat_M = x_M - residual
    out[H + gpos + 1] = (float)xM[gpos + 1] - ri;
  }
}

// ---------------------------------------------------------------------------
extern "C" void kernel_launch(void* const* d_in, const int* in_sizes, int n_in,
                              void* d_out, int out_size, void* d_ws, size_t ws_size,
                              hipStream_t stream) {
  const double* xM  = (const double*)d_in[0];   // complex128 [2048,128]
  const double* Mm  = (const double*)d_in[2];   // complex128 [256,128]
  const double* pos = (const double*)d_in[3];   // f64 [256]
  const double* doa = (const double*)d_in[4];   // f64 [4096]
  const double* gv  = (const double*)d_in[5];   // f64 [4096]
  const int*    kpt = (const int*)d_in[8];      // k = 8
  float* out = (float*)d_out;

  char* ws = (char*)d_ws;
  float*    W     = (float*)(ws);                                 // 8 MB
  float*    DM    = (float*)(ws + (size_t)(8u  << 20));           // 4 MB
  float*    norm2 = (float*)(ws + (size_t)(12u << 20));           // 512 B
  float*    Dt    = (float*)(ws + (size_t)(12u << 20) + 4096);    // 4 MB
  _Float16* Bre   = (_Float16*)(ws + (size_t)(16u << 20) + 4096); // 2 MB
  _Float16* Bim   = (_Float16*)(ws + (size_t)(18u << 20) + 4096); // 2 MB

  (void)hipMemsetAsync(norm2, 0, 128 * sizeof(float), stream);

  k_W   <<<A_GRID,          256, 0, stream>>>(pos, doa, gv, W);
  k_dm  <<<A_GRID / 8,      128, 0, stream>>>(Mm, W, DM, norm2);
  k_pack<<<M_MEAS,          256, 0, stream>>>(DM, norm2, Dt, Bre, Bim);
  k_mp  <<<N_SAMP / 16,     256, 0, stream>>>(xM, kpt, Dt, Bre, Bim, out);
}
